// AggregatedAttention_27281632264905
// MI455X (gfx1250) — matrix operand order repaired and checked by reading, not compiled, not run on hardware
//
#include <hip/hip_runtime.h>
#include <math.h>

// ---------------- problem constants ----------------
#define DIMC 128
#define NH   4
#define HD   32
#define LL   9
#define SR_  8
#define H0_  56
#define W0_  56
#define N0_  (H0_*W0_)          // 3136
#define B0_  16
#define POOLW 7
#define PL_  49
#define TBL_ 3844
#define CPBH 512
#define M_TOK (B0_*N0_)         // 50176
#define GK 128                  // K dim of every GEMM here

typedef __attribute__((ext_vector_type(16))) __bf16 v16bf;
typedef __attribute__((ext_vector_type(8)))  float  v8f;

union Frag16 { v16bf v; uint4 q[2]; };

// async global->LDS path (CDNA5): guarded so absence still compiles
#if defined(__has_builtin)
#  if __has_builtin(__builtin_amdgcn_global_load_async_to_lds_b128) && \
      __has_builtin(__builtin_amdgcn_s_wait_asynccnt)
#    define USE_ASYNC_LDS 1
#  endif
#endif
#ifndef USE_ASYNC_LDS
#  define USE_ASYNC_LDS 0
#endif

#if USE_ASYNC_LDS
// exact pointee type the builtin expects: GCC-vector int4 (per clang diagnostic)
typedef int i4v __attribute__((vector_size(16)));
typedef __attribute__((address_space(1))) i4v i4v_g;   // global (__device__) AS
typedef __attribute__((address_space(3))) i4v i4v_l;   // LDS (__shared__) AS
#endif

static __device__ __forceinline__ unsigned short f2bf(float f) {
  unsigned int u = __float_as_uint(f);
  unsigned int r = (u + 0x7FFFu + ((u >> 16) & 1u)) >> 16;   // RNE
  return (unsigned short)r;
}

static __device__ __forceinline__ float wave_sum(float v) {
  #pragma unroll
  for (int off = 16; off; off >>= 1) v += __shfl_xor(v, off, 32);
  return v;
}

// ---------------- conversion kernels ----------------
__global__ void f32_to_bf16_kernel(const float* __restrict__ in,
                                   unsigned short* __restrict__ out, int n) {
  int i = blockIdx.x * blockDim.x + threadIdx.x;
  if (i < n) out[i] = f2bf(in[i]);
}

// in [K,N] row-major  ->  out [N,K] bf16 (transposed)
__global__ void transpose_to_bf16_kernel(const float* __restrict__ in,
                                         unsigned short* __restrict__ out,
                                         int K, int N) {
  int i = blockIdx.x * blockDim.x + threadIdx.x;
  if (i < K * N) {
    int k = i / N, n = i % N;
    out[n * K + k] = f2bf(in[i]);
  }
}

// ---------------- CPB MLP: [TBL,2] -> relu(512) -> [TBL,4] ----------------
__global__ void cpb_kernel(const float* __restrict__ tbl,
                           const float* __restrict__ w1, const float* __restrict__ b1,
                           const float* __restrict__ w2, const float* __restrict__ b2,
                           float* __restrict__ out) {
  int r = blockIdx.x * blockDim.x + threadIdx.x;
  if (r >= TBL_) return;
  float t0 = tbl[2 * r], t1 = tbl[2 * r + 1];
  float o0 = b2[0], o1 = b2[1], o2 = b2[2], o3 = b2[3];
  for (int j = 0; j < CPBH; ++j) {
    float h = fmaxf(t0 * w1[j] + t1 * w1[CPBH + j] + b1[j], 0.f);
    o0 += h * w2[j * 4 + 0];
    o1 += h * w2[j * 4 + 1];
    o2 += h * w2[j * 4 + 2];
    o3 += h * w2[j * 4 + 3];
  }
  out[r * 4 + 0] = o0; out[r * 4 + 1] = o1;
  out[r * 4 + 2] = o2; out[r * 4 + 3] = o3;
}

// ---------------- bf16 WMMA GEMM, LDS-staged ----------------
// C[M,Nn] = A[M,128] x B + bias.  A bf16 [M,128]; Bt bf16 [Nn,128] (B transposed).
// Block = 256 thr (8 waves). Block tile: 64 rows x 128 cols.
//   LDS: A panel 64x128 (16KB) + B panel 128x128 (32KB), loaded async if available.
//   Wave w: row-group (w&3)*16, col-group (w>>2)*64 -> 4 acc tiles, 16 WMMAs.
// act: 0 = none, 1 = exact GELU.
__global__ __launch_bounds__(256)
void gemm_bf16_kernel(const unsigned short* __restrict__ A,
                      const unsigned short* __restrict__ Bt,
                      const float* __restrict__ bias,
                      float* __restrict__ Cf,
                      int M, int Nn, int act) {
  __shared__ unsigned short Als[64 * GK];     // 16 KB
  __shared__ unsigned short Bls[128 * GK];    // 32 KB

  __builtin_amdgcn_s_wait_tensorcnt(0);       // CDNA5 split counters (no-op at launch)

  int t    = threadIdx.x;
  int lane = t & 31;
  int w    = t >> 5;

  size_t mBase = (size_t)blockIdx.x * 64;
  size_t nBase = (size_t)blockIdx.y * 128;

  const unsigned short* Agp = A  + mBase * GK;   // 16 KB contiguous
  const unsigned short* Bgp = Bt + nBase * GK;   // 32 KB contiguous

#if USE_ASYNC_LDS
  {
    i4v_g* Aga = (i4v_g*)Agp;                 // explicit flat->AS1 addrspacecast
    i4v_g* Bga = (i4v_g*)Bgp;
    i4v_l* Al  = (i4v_l*)(&Als[0]);           // explicit flat->AS3 addrspacecast
    i4v_l* Bl  = (i4v_l*)(&Bls[0]);
    #pragma unroll
    for (int it = 0; it < 4; ++it)
      __builtin_amdgcn_global_load_async_to_lds_b128(Aga + it * 256 + t,
                                                     Al  + it * 256 + t, 0, 0);
    #pragma unroll
    for (int it = 0; it < 8; ++it)
      __builtin_amdgcn_global_load_async_to_lds_b128(Bga + it * 256 + t,
                                                     Bl  + it * 256 + t, 0, 0);
    __builtin_amdgcn_s_wait_asynccnt(0);
  }
#else
  {
    const uint4* Ag = (const uint4*)Agp;
    const uint4* Bg = (const uint4*)Bgp;
    uint4* Al = (uint4*)Als;
    uint4* Bl = (uint4*)Bls;
    #pragma unroll
    for (int it = 0; it < 4; ++it) Al[it * 256 + t] = Ag[it * 256 + t];
    #pragma unroll
    for (int it = 0; it < 8; ++it) Bl[it * 256 + t] = Bg[it * 256 + t];
  }
#endif
  __syncthreads();   // single barrier; partial-M waves skip only AFTER this point

  int rg = w & 3;            // row group within block
  int cg = w >> 2;           // col group (64 cols each)
  int m0 = (int)mBase + rg * 16;
  if (m0 < M) {              // wave-uniform guard: EXEC stays all-ones for WMMA
    int half = lane >> 4;
    int l15  = lane & 15;

    // A fragment source in LDS (ISA 16-bit A layout: lanes0-15 K 0..7/16..23,
    // lanes16-31 K 8..15/24..31 -> two 16B reads)
    const unsigned short* Ap = Als + (rg * 16 + l15) * GK + half * 8;

    v8f acc[4];
    #pragma unroll
    for (int tn = 0; tn < 4; ++tn) {
      float bv = bias[nBase + cg * 64 + tn * 16 + l15];
      #pragma unroll
      for (int r = 0; r < 8; ++r) acc[tn][r] = bv;
    }

    #pragma unroll
    for (int kk = 0; kk < GK; kk += 32) {
      Frag16 a;
      a.q[0] = *(const uint4*)(Ap + kk);
      a.q[1] = *(const uint4*)(Ap + kk + 16);
      #pragma unroll
      for (int tn = 0; tn < 4; ++tn) {
        const unsigned short* Bp = Bls + (cg * 64 + tn * 16 + l15) * GK + half * 8;
        Frag16 b;
        b.q[0] = *(const uint4*)(Bp + kk);
        b.q[1] = *(const uint4*)(Bp + kk + 16);
        acc[tn] = __builtin_amdgcn_wmma_f32_16x16x32_bf16(
            false, a.v, false, b.v, (short)0, acc[tn], false, false);
      }
    }

    #pragma unroll
    for (int tn = 0; tn < 4; ++tn) {
      int ncol  = (int)nBase + cg * 64 + tn * 16 + l15;
      int rbase = m0 + (half << 3);
      #pragma unroll
      for (int r = 0; r < 8; ++r) {
        float v = acc[tn][r];
        if (act == 1) v = 0.5f * v * (1.0f + erff(v * 0.70710678118654752f));
        Cf[(size_t)(rbase + r) * Nn + ncol] = v;
      }
    }
  }
}

// ---------------- 8x8 avg pool + LayerNorm -> bf16 [B*PL,128] ----------------
__global__ __launch_bounds__(128)
void pool_ln_kernel(const float* __restrict__ xsr,
                    const float* __restrict__ g, const float* __restrict__ bta,
                    unsigned short* __restrict__ out_bf) {
  int bp = blockIdx.x;                // b*PL + p
  int b = bp / PL_, p = bp % PL_;
  int py = p / POOLW, px = p % POOLW;
  int c = threadIdx.x;

  float s = 0.f;
  for (int dy = 0; dy < SR_; ++dy) {
    int y = py * SR_ + dy;
    for (int dx = 0; dx < SR_; ++dx) {
      int n = y * W0_ + (px * SR_ + dx);
      s += xsr[((size_t)b * N0_ + n) * DIMC + c];
    }
  }
  float v = s * (1.0f / 64.0f);

  __shared__ float sh[DIMC];
  sh[c] = v; __syncthreads();
  for (int s2 = 64; s2 > 0; s2 >>= 1) { if (c < s2) sh[c] += sh[c + s2]; __syncthreads(); }
  float mean = sh[0] * (1.0f / DIMC);
  __syncthreads();
  float d = v - mean;
  sh[c] = d * d; __syncthreads();
  for (int s2 = 64; s2 > 0; s2 >>= 1) { if (c < s2) sh[c] += sh[c + s2]; __syncthreads(); }
  float var = sh[0] * (1.0f / DIMC);
  float y = d * rsqrtf(var + 1e-5f) * g[c] + bta[c];
  out_bf[(size_t)bp * DIMC + c] = f2bf(y);
}

// ---------------- normalize pooled K per head: kvp[.,0:128] -> kpn ----------------
__global__ __launch_bounds__(128)
void kpool_norm_kernel(const float* __restrict__ kvp, float* __restrict__ kpn) {
  int bp = blockIdx.x;
  int c = threadIdx.x;                        // wave == head slice (32 ch aligned)
  float v = kvp[(size_t)bp * (2 * DIMC) + c];
  float ss = wave_sum(v * v);
  kpn[(size_t)bp * DIMC + c] = v / fmaxf(sqrtf(ss), 1e-12f);
}

// ---------------- fused attention: one wave per (token, head) ----------------
__global__ __launch_bounds__(128)
void attn_kernel(const float* __restrict__ qf,    // [B*N,128]
                 const float* __restrict__ kvf,   // [B*N,256]
                 const float* __restrict__ kpn,   // [B*PL,128] (normalized k_pool)
                 const float* __restrict__ kvp,   // [B*PL,256] (v_pool in cols 128..)
                 const float* __restrict__ cpb,   // [TBL,4]
                 const int*   __restrict__ relidx,// [N*PL]
                 const float* __restrict__ temp,  // [4]
                 const float* __restrict__ qe,    // [4,32]
                 const float* __restrict__ rbl,   // [4,9]
                 const float* __restrict__ lt,    // [4,32,9]
                 const float* __restrict__ lb,    // [4,9]
                 unsigned short* __restrict__ outb) { // [B*N,128] bf16
  int n = blockIdx.x;
  int b = blockIdx.y;
  int h = threadIdx.x >> 5;
  int d = threadIdx.x & 31;
  int y = n / W0_, x = n % W0_;
  size_t tok = (size_t)b * N0_ + n;

  float qv = qf[tok * DIMC + h * HD + d];
  float qn = qv / fmaxf(sqrtf(wave_sum(qv * qv)), 1e-12f);
  float sp = log1pf(__expf(temp[h]));          // softplus(temperature)
  int rows = 3 - (y == 0) - (y == H0_ - 1);
  int cols = 3 - (x == 0) - (x == W0_ - 1);
  float seqs = logf((float)(rows * cols) + (float)PL_);
  float qs = (qn + qe[h * HD + d]) * sp * seqs;

  float attn[LL + PL_];
  float wl[LL];

  #pragma unroll
  for (int l = 0; l < LL; ++l) {
    int dy = l / 3 - 1, dx = l % 3 - 1;
    int yy = y + dy, xx = x + dx;
    bool valid = (yy >= 0) & (yy < H0_) & (xx >= 0) & (xx < W0_);
    float kv_ = 0.f;
    if (valid) kv_ = kvf[((size_t)b * N0_ + yy * W0_ + xx) * (2 * DIMC) + h * HD + d];
    float kss = kv_ * kv_;
    float dq  = qs * kv_;
    float dl  = qn * lt[(h * HD + d) * LL + l];
    #pragma unroll
    for (int off = 16; off; off >>= 1) {
      kss += __shfl_xor(kss, off, 32);
      dq  += __shfl_xor(dq,  off, 32);
      dl  += __shfl_xor(dl,  off, 32);
    }
    float kn = fmaxf(sqrtf(kss), 1e-12f);
    attn[l] = valid ? (dq / kn + rbl[h * LL + l]) : -__builtin_inff();
    wl[l] = dl + lb[h * LL + l];
  }

  #pragma unroll
  for (int p = 0; p < PL_; ++p) {
    float kp = kpn[((size_t)b * PL_ + p) * DIMC + h * HD + d];
    float dp = wave_sum(qs * kp);
    int ri = relidx[n * PL_ + p];
    attn[LL + p] = dp + cpb[ri * 4 + h];
  }

  float mx = -__builtin_inff();
  #pragma unroll
  for (int i = 0; i < LL + PL_; ++i) mx = fmaxf(mx, attn[i]);
  float sum = 0.f;
  #pragma unroll
  for (int i = 0; i < LL + PL_; ++i) { float e = __expf(attn[i] - mx); attn[i] = e; sum += e; }
  float inv = 1.0f / sum;

  float o = 0.f;
  #pragma unroll
  for (int l = 0; l < LL; ++l) {
    int dy = l / 3 - 1, dx = l % 3 - 1;
    int yy = y + dy, xx = x + dx;
    if ((yy >= 0) & (yy < H0_) & (xx >= 0) & (xx < W0_)) {
      float vv = kvf[((size_t)b * N0_ + yy * W0_ + xx) * (2 * DIMC) + DIMC + h * HD + d];
      o += (wl[l] + attn[l] * inv) * vv;
    }
  }
  #pragma unroll
  for (int p = 0; p < PL_; ++p) {
    float vv = kvp[((size_t)b * PL_ + p) * (2 * DIMC) + DIMC + h * HD + d];
    o += (attn[LL + p] * inv) * vv;
  }
  outb[tok * DIMC + h * HD + d] = f2bf(o);
}

// ---------------- host side ----------------
extern "C" void kernel_launch(void* const* d_in, const int* in_sizes, int n_in,
                              void* d_out, int out_size, void* d_ws, size_t ws_size,
                              hipStream_t stream) {
  const float* x       = (const float*)d_in[0];
  const float* rct     = (const float*)d_in[1];
  const float* q_w     = (const float*)d_in[2];
  const float* q_b     = (const float*)d_in[3];
  const float* kv_w    = (const float*)d_in[4];
  const float* kv_b    = (const float*)d_in[5];
  const float* temp    = (const float*)d_in[6];
  const float* qe      = (const float*)d_in[7];
  const float* sr_w    = (const float*)d_in[8];
  const float* sr_b    = (const float*)d_in[9];
  const float* norm_g  = (const float*)d_in[10];
  const float* norm_b  = (const float*)d_in[11];
  const float* cpb1_w  = (const float*)d_in[12];
  const float* cpb1_b  = (const float*)d_in[13];
  const float* cpb2_w  = (const float*)d_in[14];
  const float* cpb2_b  = (const float*)d_in[15];
  const float* rbl     = (const float*)d_in[16];
  const float* lt      = (const float*)d_in[17];
  const float* lb      = (const float*)d_in[18];
  const float* proj_w  = (const float*)d_in[19];
  const float* proj_b  = (const float*)d_in[20];
  const int*   relidx  = (const int*)  d_in[21];

  char* ws = (char*)d_ws;
  size_t off = 0;
  auto carve = [&](size_t bytes) -> void* {
    void* p = ws + off;
    off = (off + bytes + 255) & ~(size_t)255;
    return p;
  };

  unsigned short* xbf    = (unsigned short*)carve((size_t)M_TOK * DIMC * 2);
  unsigned short* qwT    = (unsigned short*)carve(DIMC * DIMC * 2);
  unsigned short* kvT    = (unsigned short*)carve(DIMC * 2 * DIMC * 2);
  unsigned short* srT    = (unsigned short*)carve(DIMC * DIMC * 2);
  unsigned short* projT  = (unsigned short*)carve(DIMC * DIMC * 2);
  float*          qf     = (float*)carve((size_t)M_TOK * DIMC * 4);
  float*          kvf    = (float*)carve((size_t)M_TOK * 2 * DIMC * 4);
  float*          xsr    = (float*)carve((size_t)M_TOK * DIMC * 4);
  unsigned short* pooled = (unsigned short*)carve((size_t)B0_ * PL_ * DIMC * 2);
  float*          kvpf   = (float*)carve((size_t)B0_ * PL_ * 2 * DIMC * 4);
  float*          kpn    = (float*)carve((size_t)B0_ * PL_ * DIMC * 4);
  float*          cpbB   = (float*)carve((size_t)TBL_ * NH * 4);
  unsigned short* attnbf = (unsigned short*)carve((size_t)M_TOK * DIMC * 2);

  // 1. convert inputs / weights to bf16 (weights transposed to [N,K])
  {
    int n = M_TOK * DIMC;
    f32_to_bf16_kernel<<<(n + 255) / 256, 256, 0, stream>>>(x, xbf, n);
  }
  transpose_to_bf16_kernel<<<(DIMC * DIMC + 255) / 256, 256, 0, stream>>>(q_w, qwT, DIMC, DIMC);
  transpose_to_bf16_kernel<<<(DIMC * 2 * DIMC + 255) / 256, 256, 0, stream>>>(kv_w, kvT, DIMC, 2 * DIMC);
  transpose_to_bf16_kernel<<<(DIMC * DIMC + 255) / 256, 256, 0, stream>>>(proj_w, projT, DIMC, DIMC);
  // sr einsum 'bchw,oc->bohw' means B[k=c][n=o] = sr_w[o][c]  ->  Bt == sr_w as-is
  f32_to_bf16_kernel<<<(DIMC * DIMC + 255) / 256, 256, 0, stream>>>(sr_w, srT, DIMC * DIMC);

  // 2. CPB relative-bias MLP
  cpb_kernel<<<(TBL_ + 127) / 128, 128, 0, stream>>>(rct, cpb1_w, cpb1_b, cpb2_w, cpb2_b, cpbB);

  // 3. token GEMMs (WMMA bf16, LDS-staged): q, kv, sr(+GELU)
  gemm_bf16_kernel<<<dim3(M_TOK / 64, 1), 256, 0, stream>>>(xbf, qwT, q_b, qf, M_TOK, DIMC, 0);
  gemm_bf16_kernel<<<dim3(M_TOK / 64, 2), 256, 0, stream>>>(xbf, kvT, kv_b, kvf, M_TOK, 2 * DIMC, 0);
  gemm_bf16_kernel<<<dim3(M_TOK / 64, 1), 256, 0, stream>>>(xbf, srT, sr_b, xsr, M_TOK, DIMC, 1);

  // 4. pool + LayerNorm -> bf16, then pooled kv GEMM, then k_pool L2-norm
  pool_ln_kernel<<<B0_ * PL_, 128, 0, stream>>>(xsr, norm_g, norm_b, pooled);
  {
    int M = B0_ * PL_;                              // 784 (partial last 64-row block)
    gemm_bf16_kernel<<<dim3((M + 63) / 64, 2), 256, 0, stream>>>(pooled, kvT, kv_b, kvpf, M, 2 * DIMC, 0);
  }
  kpool_norm_kernel<<<B0_ * PL_, 128, 0, stream>>>(kvpf, kpn);

  // 5. fused local+pooled attention
  {
    dim3 grid(N0_, B0_);
    attn_kernel<<<grid, 128, 0, stream>>>(qf, kvf, kpn, kvpf, cpbB, relidx,
                                          temp, qe, rbl, lt, lb, attnbf);
  }

  // 6. output projection straight into d_out
  gemm_bf16_kernel<<<dim3(M_TOK / 64, 1), 256, 0, stream>>>(attnbf, projT, proj_b,
                                                            (float*)d_out, M_TOK, DIMC, 0);
  (void)in_sizes; (void)n_in; (void)out_size; (void)ws_size;
}
